// GENConvArchitecture_4398046511393
// MI455X (gfx1250) — matrix operand order repaired and checked by reading, not compiled, not run on hardware
//
#include <hip/hip_runtime.h>
#include <math.h>
#include <stdint.h>

// ---------------------------------------------------------------------------
// GENConv on MI455X (gfx1250, wave32).
// Edge softmax-aggregation is HBM-bound (~2.5 GB of edge traffic -> ~110us at
// 23.3 TB/s); dense node GEMMs (~11 GFLOP) are fused into WMMA f16->f32 tiles.
// Weights are pre-packed into WMMA B-fragment order so each fragment is two
// coalesced global_load_b128 per lane instead of 16 scalar u16 loads.
// ---------------------------------------------------------------------------

typedef __attribute__((ext_vector_type(16))) _Float16 v16h;
typedef __attribute__((ext_vector_type(8)))  float    v8f;

#define FIN 32
#define HD  64

// ---- WMMA fragment helpers (CDNA5 16x16x32 f16 layouts, CDNA5 ISA)

// A: 16x32 tile, row-major in LDS with row stride `stride` halves.
// lanes 0-15: row=lane, K 0..7 (elems 0..7), K 16..23 (elems 8..15)
// lanes 16-31: row=lane-16, K 8..15, K 24..31
static __device__ __forceinline__ v16h load_a_frag(const _Float16* tile, int stride,
                                                   int kblock, int lane) {
  int row  = lane & 15;
  int koff = (lane < 16) ? 0 : 8;
  const _Float16* p = tile + row * stride + kblock * 32;
  v16h a;
#pragma unroll
  for (int j = 0; j < 8; ++j) {
    a[j]     = p[koff + j];
    a[j + 8] = p[16 + koff + j];
  }
  return a;
}

// B fragments pre-packed: fragment f, lane l -> 16 contiguous halves (32B).
static __device__ __forceinline__ v16h load_b_packed(const _Float16* Wf, int frag,
                                                     int lane) {
  return ((const v16h*)Wf)[frag * 32 + lane];
}

static __device__ __forceinline__ v8f wmma_f16(v16h a, v16h b, v8f c) {
  return __builtin_amdgcn_wmma_f32_16x16x32_f16(false, a, false, b, (short)0, c,
                                                false, false);
}

static __device__ __forceinline__ float wave_sum(float v) {
#pragma unroll
  for (int o = 16; o > 0; o >>= 1) v += __shfl_xor(v, o, 32);
  return v;
}

// ---- weight packing: f32 W[K,Nout] -> f16 B-fragments -----------------------
// fragment index = kb * (Nout/16) + nb ; within fragment: lane*16 + j halves.
// lane<16: col=lane, K=kb*32+{0..7,16..23}; lane>=16: col=lane-16, K=kb*32+{8..15,24..31}.

__global__ __launch_bounds__(256) void pack_b_kernel(const float* __restrict__ W,
                                                     _Float16* __restrict__ dst,
                                                     int K, int Nout) {
  int tid = blockIdx.x * 256 + threadIdx.x;
  if (tid >= K * Nout) return;
  int j    = tid & 15;
  int lane = (tid >> 4) & 31;
  int frag = tid >> 9;  // 512 halves per fragment
  int nnb  = Nout >> 4;
  int kb   = frag / nnb;
  int nb   = frag - kb * nnb;
  int col  = lane & 15;
  int koff = (lane < 16) ? 0 : 8;
  int kk   = kb * 32 + koff + j + ((j < 8) ? 0 : 8);
  dst[tid] = (_Float16)W[kk * Nout + nb * 16 + col];
}

__global__ __launch_bounds__(256) void init_seg_kernel(float* __restrict__ m,
                                                       float* __restrict__ denom,
                                                       float* __restrict__ num, int nh) {
  int i = blockIdx.x * 256 + threadIdx.x;
  if (i >= nh) return;
  m[i] = -INFINITY;
  denom[i] = 0.f;
  num[i] = 0.f;
}

// input LayerNorm over 32 features: one wave per node, lane = feature
__global__ __launch_bounds__(256) void input_ln_kernel(const float* __restrict__ x,
                                                       const float* __restrict__ g,
                                                       const float* __restrict__ b,
                                                       float* __restrict__ h0, int n) {
  int wave = threadIdx.x >> 5, lane = threadIdx.x & 31;
  int node = blockIdx.x * 8 + wave;
  if (node >= n) return;
  float v   = x[node * FIN + lane];
  float mu  = wave_sum(v) * (1.f / 32.f);
  float dv  = v - mu;
  float var = wave_sum(dv * dv) * (1.f / 32.f);
  h0[node * FIN + lane] = dv * rsqrtf(var + 1e-5f) * g[lane] + b[lane];
}

// ---- edge passes ----------------------------------------------------------

static __device__ __forceinline__ void atomicMaxF(float* a, float v) {
  if (v >= 0.f) atomicMax((int*)a, __float_as_int(v));
  else          atomicMin((unsigned int*)a, __float_as_uint(v));
}

// pass A: segment max of logits = (relu(h_src[src]) + eps) * t  into m[dst][c]
// one wave per edge (uniform edge index -> scalar s_loads of src/dst),
// 2 channels per lane via float2.
__global__ __launch_bounds__(256) void edge_max_kernel(const int* __restrict__ src,
                                                       const int* __restrict__ dst,
                                                       const float* __restrict__ hsrc,
                                                       const float* __restrict__ tptr,
                                                       float* __restrict__ m, int E) {
  int e = __builtin_amdgcn_readfirstlane(blockIdx.x * 8 + (threadIdx.x >> 5));
  if (e >= E) return;
  int c2 = threadIdx.x & 31;
  int s = src[e], d = dst[e];
  float tv = tptr[0];
  float2 v = ((const float2*)(hsrc + (size_t)s * HD))[c2];
  float l0 = (fmaxf(v.x, 0.f) + 1e-7f) * tv;
  float l1 = (fmaxf(v.y, 0.f) + 1e-7f) * tv;
  float* mp = m + (size_t)d * HD + c2 * 2;
  atomicMaxF(mp, l0);
  atomicMaxF(mp + 1, l1);
}

// pass B: e = exp(logit - m[dst]); denom += e; num += msg*e  (agg = num/(denom+eps))
__global__ __launch_bounds__(256) void edge_sum_kernel(const int* __restrict__ src,
                                                       const int* __restrict__ dst,
                                                       const float* __restrict__ hsrc,
                                                       const float* __restrict__ tptr,
                                                       const float* __restrict__ m,
                                                       float* __restrict__ denom,
                                                       float* __restrict__ num, int E) {
  int e = __builtin_amdgcn_readfirstlane(blockIdx.x * 8 + (threadIdx.x >> 5));
  if (e >= E) return;
  int c2 = threadIdx.x & 31;
  int s = src[e], d = dst[e];
  float tv = tptr[0];
  float2 v = ((const float2*)(hsrc + (size_t)s * HD))[c2];
  float msg0 = fmaxf(v.x, 0.f) + 1e-7f;
  float msg1 = fmaxf(v.y, 0.f) + 1e-7f;
  size_t base = (size_t)d * HD + c2 * 2;
  float e0 = expf(msg0 * tv - m[base]);
  float e1 = expf(msg1 * tv - m[base + 1]);
  atomicAdd(&denom[base], e0);
  atomicAdd(&denom[base + 1], e1);
  atomicAdd(&num[base], msg0 * e0);
  atomicAdd(&num[base + 1], msg1 * e1);
}

// ---- layer-0 projection: out[N,64] = A[N,32] @ W(f16 frags)[32,64] + bias --

__global__ __launch_bounds__(256) void proj_kernel(const float* __restrict__ A,
                                                   const _Float16* __restrict__ Wf,
                                                   const float* __restrict__ bias,
                                                   float* __restrict__ out, int n) {
  __shared__ _Float16 s_a[8][16][32];
  int wave = threadIdx.x >> 5, lane = threadIdx.x & 31;
  int row0 = blockIdx.x * 128 + wave * 16;
  for (int i = lane; i < 16 * 32; i += 32) {
    int r = i >> 5, c = i & 31;
    int gr = row0 + r; if (gr >= n) gr = n - 1;
    s_a[wave][r][c] = (_Float16)A[(size_t)gr * FIN + c];
  }
  __syncthreads();
  int col = lane & 15;
  int rb = (lane & 16) ? 8 : 0;
  bool full = (row0 + 16 <= n);
  v16h a = load_a_frag(&s_a[wave][0][0], 32, 0, lane);
#pragma unroll
  for (int nb = 0; nb < 4; ++nb) {
    v8f acc = {};
    v16h b = load_b_packed(Wf, nb, lane);  // K=32 -> single k-block
    acc = wmma_f16(a, b, acc);
    float bv = bias[nb * 16 + col];
    float* op = out + (size_t)(row0 + rb) * HD + nb * 16 + col;
    if (full) {
#pragma unroll
      for (int r = 0; r < 8; ++r) op[(size_t)r * HD] = acc[r] + bv;
    } else {
#pragma unroll
      for (int r = 0; r < 8; ++r)
        if (row0 + rb + r < n) op[(size_t)r * HD] = acc[r] + bv;
    }
  }
}

// ---- fused node update: a = num/(denom+eps)+xdst; GEMM1+b1 -> LN128 -> ReLU
// ---- -> GEMM2+b2 -> LN64 (+optional ReLU) -> out. 4 waves, 16 rows/wave ----

__global__ __launch_bounds__(128) void update_kernel(
    const float* __restrict__ num, const float* __restrict__ denom,
    const float* __restrict__ xdst,
    const _Float16* __restrict__ W1f, const float* __restrict__ b1,
    const float* __restrict__ lng, const float* __restrict__ lnb,
    const _Float16* __restrict__ W2f, const float* __restrict__ b2,
    const float* __restrict__ ng, const float* __restrict__ nbv,
    float* __restrict__ out, int n, int relu_out) {
  __shared__ _Float16 s_h[4][16][128];  // A input (cols 0..63) then mid (128)
  __shared__ float    s_f[4][16][128];  // f32 staging for LayerNorm
  int wave = threadIdx.x >> 5, lane = threadIdx.x & 31;
  int row0 = blockIdx.x * 64 + wave * 16;
  int col = lane & 15;
  int rb = (lane & 16) ? 8 : 0;
  bool full = (row0 + 16 <= n);

  // stage 1: a = num/(denom+1e-16) + xdst  -> f16 in s_h[..][0..63]
  for (int i = lane; i < 16 * HD; i += 32) {
    int r = i >> 6, c = i & 63;
    int gr = row0 + r; if (gr >= n) gr = n - 1;
    size_t idx = (size_t)gr * HD + c;
    float a = num[idx] / (denom[idx] + 1e-16f) + xdst[idx];
    s_h[wave][r][c] = (_Float16)a;
  }
  __syncthreads();

  // GEMM1: [16x64] @ W1[64,128] + b1 -> s_f   (frag = kb*8 + nb)
#pragma unroll
  for (int nb = 0; nb < 8; ++nb) {
    v8f acc = {};
#pragma unroll
    for (int kb = 0; kb < 2; ++kb) {
      v16h a = load_a_frag(&s_h[wave][0][0], 128, kb, lane);
      v16h b = load_b_packed(W1f, kb * 8 + nb, lane);
      acc = wmma_f16(a, b, acc);
    }
    float bv = b1[nb * 16 + col];
#pragma unroll
    for (int r = 0; r < 8; ++r) s_f[wave][rb + r][nb * 16 + col] = acc[r] + bv;
  }
  __syncthreads();

  // LayerNorm over 128 + ReLU -> f16 mid in s_h
  for (int r = 0; r < 16; ++r) {
    float s = 0.f, s2 = 0.f;
#pragma unroll
    for (int q = 0; q < 4; ++q) {
      float v = s_f[wave][r][lane + 32 * q];
      s += v; s2 += v * v;
    }
    s = wave_sum(s); s2 = wave_sum(s2);
    float mu  = s * (1.f / 128.f);
    float var = s2 * (1.f / 128.f) - mu * mu;
    float rs  = rsqrtf(var + 1e-5f);
#pragma unroll
    for (int q = 0; q < 4; ++q) {
      int c = lane + 32 * q;
      float v = (s_f[wave][r][c] - mu) * rs * lng[c] + lnb[c];
      s_h[wave][r][c] = (_Float16)fmaxf(v, 0.f);
    }
  }
  __syncthreads();

  // GEMM2: [16x128] @ W2[128,64] + b2 -> s_f cols 0..63   (frag = kb*4 + nb)
#pragma unroll
  for (int nb = 0; nb < 4; ++nb) {
    v8f acc = {};
#pragma unroll
    for (int kb = 0; kb < 4; ++kb) {
      v16h a = load_a_frag(&s_h[wave][0][0], 128, kb, lane);
      v16h b = load_b_packed(W2f, kb * 4 + nb, lane);
      acc = wmma_f16(a, b, acc);
    }
    float bv = b2[nb * 16 + col];
#pragma unroll
    for (int r = 0; r < 8; ++r) s_f[wave][rb + r][nb * 16 + col] = acc[r] + bv;
  }
  __syncthreads();

  // outer LayerNorm over 64 (+optional ReLU), store
  float g0 = ng[lane], g1 = ng[lane + 32];
  float bb0 = nbv[lane], bb1 = nbv[lane + 32];
  for (int r = 0; r < 16; ++r) {
    float v0 = s_f[wave][r][lane];
    float v1 = s_f[wave][r][lane + 32];
    float s  = wave_sum(v0 + v1);
    float s2 = wave_sum(v0 * v0 + v1 * v1);
    float mu  = s * (1.f / 64.f);
    float var = s2 * (1.f / 64.f) - mu * mu;
    float rs  = rsqrtf(var + 1e-5f);
    float o0 = (v0 - mu) * rs * g0 + bb0;
    float o1 = (v1 - mu) * rs * g1 + bb1;
    if (relu_out) { o0 = fmaxf(o0, 0.f); o1 = fmaxf(o1, 0.f); }
    int gr = row0 + r;
    if (full || gr < n) {
      out[(size_t)gr * HD + lane]      = o0;
      out[(size_t)gr * HD + lane + 32] = o1;
    }
  }
}

// ---- heads: two MLPs 64->32->1, one wave per node -------------------------

__global__ __launch_bounds__(256) void head_kernel(
    const float* __restrict__ h,
    const float* __restrict__ W1a, const float* __restrict__ b1a,
    const float* __restrict__ W2a, const float* __restrict__ b2a,
    const float* __restrict__ W1b, const float* __restrict__ b1b,
    const float* __restrict__ W2b, const float* __restrict__ b2b,
    float* __restrict__ out, int n) {
  int wave = threadIdx.x >> 5, lane = threadIdx.x & 31;
  int node = blockIdx.x * 8 + wave;
  if (node >= n) return;
  const float* hr = h + (size_t)node * HD;
  float h0v = hr[lane], h1v = hr[lane + 32];
#pragma unroll
  for (int head = 0; head < 2; ++head) {
    const float* W1 = head ? W1b : W1a;
    const float* b1 = head ? b1b : b1a;
    const float* W2 = head ? W2b : W2a;
    const float* b2 = head ? b2b : b2a;
    float z = b1[lane];
    for (int k = 0; k < 32; ++k) z += __shfl(h0v, k, 32) * W1[k * 32 + lane];
    for (int k = 0; k < 32; ++k) z += __shfl(h1v, k, 32) * W1[(k + 32) * 32 + lane];
    z = fmaxf(z, 0.f);
    float p = wave_sum(z * W2[lane]);
    if (lane == 0) out[(size_t)node * 2 + head] = p + b2[0];
  }
}

// ---------------------------------------------------------------------------
// host launch
// d_in order (setup_inputs insertion order, leaves recursively):
//  0 x  1 edge_index  2 in_g  3 in_b
//  layer0 @4: t W1 b1 ln_g ln_b W2 b2 norm_g norm_b Wsrc bsrc Wdst bdst
//  layer1 @17, layer2 @26: t W1 b1 ln_g ln_b W2 b2 norm_g norm_b
//  rtt @35: W1 b1 W2 b2 ; retrans @39: W1 b1 W2 b2
// ---------------------------------------------------------------------------

extern "C" void kernel_launch(void* const* d_in, const int* in_sizes, int n_in,
                              void* d_out, int out_size, void* d_ws, size_t ws_size,
                              hipStream_t stream) {
  const int N = in_sizes[0] / FIN;
  const int E = in_sizes[1] / 2;
  const float* x   = (const float*)d_in[0];
  const int*   ei  = (const int*)d_in[1];
  const int* srcI  = ei;
  const int* dstI  = ei + E;
  const float* ing = (const float*)d_in[2];
  const float* inb = (const float*)d_in[3];
  const int LB[3] = {4, 17, 26};

  float* ws = (float*)d_ws;
  size_t off = 0;
  float* h0    = ws + off; off += (size_t)N * FIN;
  float* hsrc0 = ws + off; off += (size_t)N * HD;
  float* xdst0 = ws + off; off += (size_t)N * HD;
  float* hA    = ws + off; off += (size_t)N * HD;
  float* hB    = ws + off; off += (size_t)N * HD;
  float* mbuf  = ws + off; off += (size_t)N * HD;
  float* dbuf  = ws + off; off += (size_t)N * HD;
  float* nbuf  = ws + off; off += (size_t)N * HD;

  uintptr_t hp = ((uintptr_t)(ws + off) + 31) & ~(uintptr_t)31;  // 32B align v16h
  _Float16* hf = (_Float16*)hp;
  size_t ho = 0;
  _Float16* Wsrc16 = hf + ho; ho += FIN * HD;
  _Float16* Wdst16 = hf + ho; ho += FIN * HD;
  _Float16* W1h[3]; _Float16* W2h[3];
  for (int i = 0; i < 3; ++i) {
    W1h[i] = hf + ho; ho += HD * 2 * HD;
    W2h[i] = hf + ho; ho += 2 * HD * HD;
  }

  // pack weights into WMMA B-fragment order (deterministic each launch)
  pack_b_kernel<<<(FIN * HD + 255) / 256, 256, 0, stream>>>((const float*)d_in[13], Wsrc16, FIN, HD);
  pack_b_kernel<<<(FIN * HD + 255) / 256, 256, 0, stream>>>((const float*)d_in[15], Wdst16, FIN, HD);
  for (int i = 0; i < 3; ++i) {
    pack_b_kernel<<<(HD * 2 * HD + 255) / 256, 256, 0, stream>>>((const float*)d_in[LB[i] + 1], W1h[i], HD, 2 * HD);
    pack_b_kernel<<<(2 * HD * HD + 255) / 256, 256, 0, stream>>>((const float*)d_in[LB[i] + 5], W2h[i], 2 * HD, HD);
  }

  input_ln_kernel<<<(N + 7) / 8, 256, 0, stream>>>(x, ing, inb, h0, N);

  float* houts[3] = {hA, hB, hA};
  const int egrid = (E + 7) / 8;  // one wave (32 lanes) per edge, 8 edges/block
  for (int i = 0; i < 3; ++i) {
    const float* hsrc;
    const float* xdst;
    if (i == 0) {
      proj_kernel<<<(N + 127) / 128, 256, 0, stream>>>(h0, Wsrc16, (const float*)d_in[14], hsrc0, N);
      proj_kernel<<<(N + 127) / 128, 256, 0, stream>>>(h0, Wdst16, (const float*)d_in[16], xdst0, N);
      hsrc = hsrc0; xdst = xdst0;
    } else {
      hsrc = (i == 1) ? hA : hB;
      xdst = hsrc;
    }
    init_seg_kernel<<<((size_t)N * HD + 255) / 256, 256, 0, stream>>>(mbuf, dbuf, nbuf, N * HD);
    edge_max_kernel<<<egrid, 256, 0, stream>>>(srcI, dstI, hsrc, (const float*)d_in[LB[i]], mbuf, E);
    edge_sum_kernel<<<egrid, 256, 0, stream>>>(srcI, dstI, hsrc, (const float*)d_in[LB[i]], mbuf, dbuf, nbuf, E);
    update_kernel<<<(N + 63) / 64, 128, 0, stream>>>(
        nbuf, dbuf, xdst, W1h[i], (const float*)d_in[LB[i] + 2],
        (const float*)d_in[LB[i] + 3], (const float*)d_in[LB[i] + 4],
        W2h[i], (const float*)d_in[LB[i] + 6],
        (const float*)d_in[LB[i] + 7], (const float*)d_in[LB[i] + 8],
        houts[i], N, (i < 2) ? 1 : 0);
  }

  head_kernel<<<(N + 7) / 8, 256, 0, stream>>>(
      hA,
      (const float*)d_in[35], (const float*)d_in[36], (const float*)d_in[37], (const float*)d_in[38],
      (const float*)d_in[39], (const float*)d_in[40], (const float*)d_in[41], (const float*)d_in[42],
      (float*)d_out, N);
}
